// MMPool_18657337933845
// MI455X (gfx1250) — matrix operand order repaired and checked by reading, not compiled, not run on hardware
//
#include <hip/hip_runtime.h>
#include <math.h>

// ---------------- problem constants (from reference) ----------------
#define RR     273
#define BB     1024
#define CC     128
#define DEG    32
#define NN     (BB * RR)          // 279552 nodes
#define EE     (NN * DEG)         // 8945664 edges per modality
#define KK     219                // ceil(0.8 * 273)
#define BK     (BB * KK)          // 224256 kept nodes
#define NB_E   ((EE + 255) / 256) // 34944 edge blocks

typedef __attribute__((ext_vector_type(16))) _Float16 v16h;
typedef __attribute__((ext_vector_type(8)))  float    v8f;

// ---------------------------------------------------------------------------
// 1) degree init (self-loop contributes 1)
__global__ __launch_bounds__(256) void mmp_deg_init(float* deg) {
  int i = blockIdx.x * 256 + threadIdx.x;
  if (i < NN) deg[i] = 1.0f;
}

// 2) degree histogram over targets
__global__ __launch_bounds__(256) void mmp_deg_count(const int* __restrict__ ei,
                                                     float* __restrict__ deg) {
  int i = blockIdx.x * 256 + threadIdx.x;
  if (i < EE) atomicAdd(&deg[ei[EE + i]], 1.0f);
}

// 3) deg -> D^-1/2 in place
__global__ __launch_bounds__(256) void mmp_dinv(float* deg) {
  int i = blockIdx.x * 256 + threadIdx.x;
  if (i < NN) deg[i] = rsqrtf(fmaxf(deg[i], 1.0f));
}

// ---------------------------------------------------------------------------
// 4) xw = X @ W  via WMMA f32_16x16x32_f16.
//    Each wave owns a 16-row tile. A: 16x32 f16 per k-chunk (ISA 16-bit A layout:
//    lanes 0-15 row M, K {0..7,16..23}; lanes 16-31 row M, K {8..15,24..31}).
//    B: W in column 0 only (lane 0 carries K {0..7,16..23}, lane 16 carries K {8..15,24..31}).
//    Result column 0: lane 0 VGPR m = row m, lane 16 VGPR m = row 8+m.
__global__ __launch_bounds__(256) void mmp_gemv_wmma(const float* __restrict__ x,
                                                     const float* __restrict__ W,
                                                     const float* __restrict__ bias_p,
                                                     float* __restrict__ xw) {
  int wave = (blockIdx.x * blockDim.x + threadIdx.x) >> 5;
  int lane = threadIdx.x & 31;
  int row0 = wave * 16;
  if (row0 >= NN) return;                 // wave-uniform guard (EXEC stays all-1)

  int  lrow = lane & 15;
  bool hi   = lane >= 16;
  const float* xrow = x + (size_t)(row0 + lrow) * CC;

  v8f c = {0.f, 0.f, 0.f, 0.f, 0.f, 0.f, 0.f, 0.f};

  for (int kc = 0; kc < CC; kc += 32) {
    v16h a, b;
    int k0 = kc + (hi ? 8 : 0);
    int k1 = kc + (hi ? 24 : 16);
#pragma unroll
    for (int j = 0; j < 8; ++j) a[j]     = (_Float16)xrow[k0 + j];
#pragma unroll
    for (int j = 0; j < 8; ++j) a[8 + j] = (_Float16)xrow[k1 + j];
#pragma unroll
    for (int j = 0; j < 16; ++j) b[j] = (_Float16)0.f;
    if (lane == 0 || lane == 16) {
#pragma unroll
      for (int j = 0; j < 8; ++j) b[j]     = (_Float16)W[k0 + j];
#pragma unroll
      for (int j = 0; j < 8; ++j) b[8 + j] = (_Float16)W[k1 + j];
    }
    c = __builtin_amdgcn_wmma_f32_16x16x32_f16(false, a, false, b,
                                               (short)0, c, false, false);
  }

  float bias = bias_p[0];
  if (lane == 0) {
#pragma unroll
    for (int m = 0; m < 8; ++m) xw[row0 + m] = c[m] + bias;
  } else if (lane == 16) {
#pragma unroll
    for (int m = 0; m < 8; ++m) xw[row0 + 8 + m] = c[m] + bias;
  }
}

// ---------------------------------------------------------------------------
// 5) acc[i] = dinv[i]^2 * xw[i]  (self-loop term)
__global__ __launch_bounds__(256) void mmp_acc_init(const float* __restrict__ dinv,
                                                    const float* __restrict__ xw,
                                                    float* __restrict__ acc) {
  int i = blockIdx.x * 256 + threadIdx.x;
  if (i < NN) { float d = dinv[i]; acc[i] = d * d * xw[i]; }
}

// 6) edge scatter: acc[t] += dinv[s]*dinv[t]*xw[s]
__global__ __launch_bounds__(256) void mmp_scatter(const int* __restrict__ ei,
                                                   const float* __restrict__ dinv,
                                                   const float* __restrict__ xw,
                                                   float* __restrict__ acc) {
  int i = blockIdx.x * 256 + threadIdx.x;
  if (i < EE) {
    int s = ei[i], t = ei[EE + i];
    atomicAdd(&acc[t], dinv[s] * dinv[t] * xw[s]);
  }
}

// 7) combine: score = gw0*tanh(a1) + gw1*tanh(a2) + gb
__global__ __launch_bounds__(256) void mmp_combine(const float* __restrict__ acc1,
                                                   const float* __restrict__ acc2,
                                                   const float* __restrict__ gw,
                                                   const float* __restrict__ gb,
                                                   float* __restrict__ score_ws,
                                                   float* __restrict__ out_score,
                                                   float* __restrict__ out_s1,
                                                   float* __restrict__ out_s2) {
  int i = blockIdx.x * 256 + threadIdx.x;
  if (i < NN) {
    float s1 = tanhf(acc1[i]);
    float s2 = tanhf(acc2[i]);
    float sc = gw[0] * s1 + gw[1] * s2 + gb[0];
    score_ws[i] = sc;
    out_score[i] = sc;
    out_s1[i] = s1;
    out_s2[i] = s2;
  }
}

// ---------------------------------------------------------------------------
// 8) exact per-graph top-K via rank counting (descending, stable tie-break:
//    rank = #{j : s[j] > s[i]  or  (s[j]==s[i] && j < i)}  matches lax.top_k)
__global__ __launch_bounds__(288) void mmp_topk(const float* __restrict__ score,
                                                int* __restrict__ new_id,
                                                int* __restrict__ perm_i,
                                                float* __restrict__ gate,
                                                float* __restrict__ out_perm,
                                                float* __restrict__ out_batch) {
  __shared__ float s[RR];
  int b = blockIdx.x;
  int t = threadIdx.x;
  if (t < RR) s[t] = score[b * RR + t];
  __syncthreads();
  if (t >= RR) return;
  float mine = s[t];
  int rank = 0;
  for (int j = 0; j < RR; ++j) {
    float v = s[j];
    rank += (v > mine) || (v == mine && j < t);
  }
  int g = b * RR + t;
  if (rank < KK) {
    int o = b * KK + rank;
    new_id[g]  = o;
    perm_i[o]  = g;
    gate[o]    = tanhf(mine);
    out_perm[o]  = (float)g;
    out_batch[o] = (float)b;
  } else {
    new_id[g] = -1;
  }
}

// 9) gated gather: out[j] = x[perm[j]] * gate[j]   (float4 vectorized)
__global__ __launch_bounds__(256) void mmp_gather(const float* __restrict__ x,
                                                  const int* __restrict__ perm_i,
                                                  const float* __restrict__ gate,
                                                  float* __restrict__ out) {
  int i = blockIdx.x * 256 + threadIdx.x;      // over BK * (CC/4)
  if (i >= BK * (CC / 4)) return;
  int j  = i >> 5;                             // output row
  int c4 = i & 31;                             // float4 slot
  float g = gate[j];
  const float4* src = (const float4*)(x + (size_t)perm_i[j] * CC);
  float4 v = src[c4];
  v.x *= g; v.y *= g; v.z *= g; v.w *= g;
  ((float4*)(out + (size_t)j * CC))[c4] = v;
}

// ---------------------------------------------------------------------------
// 10) filter_adj pass A: per-block valid-edge counts (wave32 ballot reduce)
__global__ __launch_bounds__(256) void mmp_edge_count(const int* __restrict__ ei,
                                                      const int* __restrict__ new_id,
                                                      int* __restrict__ bsum) {
  int i = blockIdx.x * 256 + threadIdx.x;
  int valid = 0;
  if (i < EE) valid = (new_id[ei[i]] >= 0) && (new_id[ei[EE + i]] >= 0);
  unsigned m = (unsigned)__ballot(valid);      // wave32: low 32 bits
  int wid = threadIdx.x >> 5, lane = threadIdx.x & 31;
  __shared__ int wsum[8];
  if (lane == 0) wsum[wid] = __popc(m);
  __syncthreads();
  if (threadIdx.x == 0) {
    int s = 0;
    for (int w = 0; w < 8; ++w) s += wsum[w];
    bsum[blockIdx.x] = s;
  }
}

// 11) single-block exclusive scan of NB_E block counts (+ grand total)
__global__ __launch_bounds__(1024) void mmp_scan(int* __restrict__ bsum,
                                                 int* __restrict__ total) {
  __shared__ int tmp[1024];
  __shared__ int carry;
  if (threadIdx.x == 0) carry = 0;
  __syncthreads();
  for (int base = 0; base < NB_E; base += 1024) {
    int i = base + threadIdx.x;
    int v = (i < NB_E) ? bsum[i] : 0;
    tmp[threadIdx.x] = v;
    __syncthreads();
    for (int ofs = 1; ofs < 1024; ofs <<= 1) {
      int t = (threadIdx.x >= ofs) ? tmp[threadIdx.x - ofs] : 0;
      __syncthreads();
      tmp[threadIdx.x] += t;
      __syncthreads();
    }
    int incl = tmp[threadIdx.x];
    int excl = incl - v + carry;
    if (i < NB_E) bsum[i] = excl;
    __syncthreads();
    if (threadIdx.x == 1023) carry += incl;   // incl of last lane == chunk total
    __syncthreads();
  }
  if (threadIdx.x == 0) *total = carry;
}

// 12) filter_adj pass C: stable partition scatter.
//     valid edge  -> pos = excl_valid_prefix(i)
//     invalid     -> pos = V_total + (i - excl_valid_prefix(i)), payload (-1,-1,0)
__global__ __launch_bounds__(256) void mmp_edge_scatter(const int* __restrict__ ei,
                                                        const float* __restrict__ ea,
                                                        const int* __restrict__ new_id,
                                                        const int* __restrict__ bofs,
                                                        const int* __restrict__ total,
                                                        float* __restrict__ out_ei,
                                                        float* __restrict__ out_ea) {
  int i = blockIdx.x * 256 + threadIdx.x;
  int wid = threadIdx.x >> 5, lane = threadIdx.x & 31;
  int ns = -1, nt = -1, valid = 0;
  float a = 0.f;
  if (i < EE) {
    ns = new_id[ei[i]];
    nt = new_id[ei[EE + i]];
    valid = (ns >= 0) && (nt >= 0);
    a = ea[i];
  }
  unsigned m = (unsigned)__ballot(valid);
  int lpre = __popc(m & ((1u << lane) - 1u));
  __shared__ int wcnt[8];
  __shared__ int wexc[8];
  if (lane == 0) wcnt[wid] = __popc(m);
  __syncthreads();
  if (threadIdx.x == 0) {
    int s = 0;
    for (int w = 0; w < 8; ++w) { wexc[w] = s; s += wcnt[w]; }
  }
  __syncthreads();
  if (i < EE) {
    int vpre = bofs[blockIdx.x] + wexc[wid] + lpre;  // global excl prefix of valid
    int pos;
    if (valid) {
      pos = vpre;
    } else {
      pos = *total + (i - vpre);
      ns = -1; nt = -1; a = 0.f;
    }
    out_ei[pos]      = (float)ns;
    out_ei[EE + pos] = (float)nt;
    out_ea[pos]      = a;
  }
}

// ---------------------------------------------------------------------------
extern "C" void kernel_launch(void* const* d_in, const int* in_sizes, int n_in,
                              void* d_out, int out_size, void* d_ws, size_t ws_size,
                              hipStream_t stream) {
  (void)in_sizes; (void)n_in; (void)out_size; (void)ws_size;

  const float* x1  = (const float*)d_in[0];
  const float* x2  = (const float*)d_in[1];
  const int*   ei1 = (const int*)  d_in[2];
  const int*   ei2 = (const int*)  d_in[3];
  const float* ea1 = (const float*)d_in[4];
  const float* ea2 = (const float*)d_in[5];
  /* batch = d_in[6] (unused: batch_o derivable) */
  const float* W1  = (const float*)d_in[7];
  const float* b1  = (const float*)d_in[8];
  const float* W2  = (const float*)d_in[9];
  const float* b2  = (const float*)d_in[10];
  const float* gw  = (const float*)d_in[11];
  const float* gb  = (const float*)d_in[12];

  // ---- output layout (flat, reference return order) ----
  float* out = (float*)d_out;
  size_t o = 0;
  float* o_x1    = out + o; o += (size_t)BK * CC;
  float* o_x2    = out + o; o += (size_t)BK * CC;
  float* o_ei1   = out + o; o += (size_t)2 * EE;
  float* o_ei2   = out + o; o += (size_t)2 * EE;
  float* o_ea1   = out + o; o += (size_t)EE;
  float* o_ea2   = out + o; o += (size_t)EE;
  float* o_batch = out + o; o += (size_t)BK;
  float* o_perm  = out + o; o += (size_t)BK;
  float* o_score = out + o; o += (size_t)NN;
  float* o_s1    = out + o; o += (size_t)NN;
  float* o_s2    = out + o; o += (size_t)NN;

  // ---- workspace carve ----
  char* w = (char*)d_ws;
  float* dinv1  = (float*)w; w += sizeof(float) * NN;
  float* dinv2  = (float*)w; w += sizeof(float) * NN;
  float* xw1    = (float*)w; w += sizeof(float) * NN;
  float* xw2    = (float*)w; w += sizeof(float) * NN;
  float* acc1   = (float*)w; w += sizeof(float) * NN;
  float* acc2   = (float*)w; w += sizeof(float) * NN;
  float* sc_ws  = (float*)w; w += sizeof(float) * NN;
  int*   new_id = (int*)  w; w += sizeof(int)   * NN;
  int*   perm_i = (int*)  w; w += sizeof(int)   * BK;
  float* gate   = (float*)w; w += sizeof(float) * BK;
  int*   bsum   = (int*)  w; w += sizeof(int)   * NB_E;
  int*   total  = (int*)  w; w += sizeof(int);

  const int gN  = (NN + 255) / 256;         // 1092
  const int gE  = NB_E;                     // 34944
  const int gW  = (NN / 16 * 32 + 255) / 256; // 2184 (16 rows per wave)
  const int gG  = (BK * (CC / 4) + 255) / 256;

  // degrees -> D^-1/2
  mmp_deg_init<<<gN, 256, 0, stream>>>(dinv1);
  mmp_deg_init<<<gN, 256, 0, stream>>>(dinv2);
  mmp_deg_count<<<gE, 256, 0, stream>>>(ei1, dinv1);
  mmp_deg_count<<<gE, 256, 0, stream>>>(ei2, dinv2);
  mmp_dinv<<<gN, 256, 0, stream>>>(dinv1);
  mmp_dinv<<<gN, 256, 0, stream>>>(dinv2);

  // xw = X @ W + b   (WMMA)
  mmp_gemv_wmma<<<gW, 256, 0, stream>>>(x1, W1, b1, xw1);
  mmp_gemv_wmma<<<gW, 256, 0, stream>>>(x2, W2, b2, xw2);

  // GCN propagate: self loop + edge scatter
  mmp_acc_init<<<gN, 256, 0, stream>>>(dinv1, xw1, acc1);
  mmp_acc_init<<<gN, 256, 0, stream>>>(dinv2, xw2, acc2);
  mmp_scatter<<<gE, 256, 0, stream>>>(ei1, dinv1, xw1, acc1);
  mmp_scatter<<<gE, 256, 0, stream>>>(ei2, dinv2, xw2, acc2);

  // fuse scores
  mmp_combine<<<gN, 256, 0, stream>>>(acc1, acc2, gw, gb, sc_ws,
                                      o_score, o_s1, o_s2);

  // per-graph top-K + relabel map
  mmp_topk<<<BB, 288, 0, stream>>>(sc_ws, new_id, perm_i, gate, o_perm, o_batch);

  // gated node gather
  mmp_gather<<<gG, 256, 0, stream>>>(x1, perm_i, gate, o_x1);
  mmp_gather<<<gG, 256, 0, stream>>>(x2, perm_i, gate, o_x2);

  // stable edge filter, modality 1 then 2 (stream-ordered buffer reuse)
  mmp_edge_count<<<gE, 256, 0, stream>>>(ei1, new_id, bsum);
  mmp_scan<<<1, 1024, 0, stream>>>(bsum, total);
  mmp_edge_scatter<<<gE, 256, 0, stream>>>(ei1, ea1, new_id, bsum, total, o_ei1, o_ea1);

  mmp_edge_count<<<gE, 256, 0, stream>>>(ei2, new_id, bsum);
  mmp_scan<<<1, 1024, 0, stream>>>(bsum, total);
  mmp_edge_scatter<<<gE, 256, 0, stream>>>(ei2, ea2, new_id, bsum, total, o_ei2, o_ea2);
}
